// LSTM_Decoder_55482387529792
// MI455X (gfx1250) — compile-verified, matrix operand order checked by reading, not run on hardware
//
#include <hip/hip_runtime.h>
#include <hip/hip_bf16.h>

// ---------------- problem constants (match reference) ----------------
constexpr int kB = 32;      // batch
constexpr int kT = 64;      // time steps
constexpr int kE = 512;     // embed dim
constexpr int kH = 1024;    // hidden
constexpr int kV = 32000;   // vocab
constexpr int k4H = 4 * kH; // gate dim

typedef __bf16 bf16;
typedef __attribute__((ext_vector_type(16))) __bf16 v16bf;
typedef __attribute__((ext_vector_type(8)))  float  v8f;

// 16-bit A/B fragment: 16 bf16 = 8 VGPRs per lane.
// Layout (CDNA5 ISA, 16-bit A 16x32): lane<16 holds K {0..7,16..23},
// lane>=16 holds K {8..15,24..31} -> two contiguous 16B loads per fragment.
union FragBF {
    v16bf v;
    uint4 q[2];
};

__device__ __forceinline__ v8f wmma_bf16(const FragBF& a, const FragBF& b, v8f c) {
    return __builtin_amdgcn_wmma_f32_16x16x32_bf16(
        /*neg_a=*/false, a.v, /*neg_b=*/false, b.v,
        /*c_mod=*/(short)0, c, /*reuse_a=*/false, /*reuse_b=*/false);
}

__device__ __forceinline__ float sigmoidf_(float x) {
    return 1.0f / (1.0f + __expf(-x));
}

__device__ __forceinline__ v8f vzero8() {
    v8f z = {0.f, 0.f, 0.f, 0.f, 0.f, 0.f, 0.f, 0.f};
    return z;
}

// ---- CDNA5 async global->LDS copy (ASYNCcnt-tracked), 16B per lane ----
__device__ __forceinline__ void async_copy_b128(unsigned lds_byte_addr, const void* gsrc) {
    asm volatile("global_load_async_to_lds_b128 %0, %1, off"
                 :: "v"(lds_byte_addr), "v"(gsrc) : "memory");
}
__device__ __forceinline__ void wait_asynccnt0() {
#if __has_builtin(__builtin_amdgcn_s_wait_asynccnt)
    __builtin_amdgcn_s_wait_asynccnt(0);
#else
    asm volatile("s_wait_asynccnt 0x0" ::: "memory");
#endif
}
__device__ __forceinline__ unsigned lds_addr_of(const void* p) {
    return (unsigned)(size_t)p;   // addrspace(3) offset lives in low 32 bits
}

// ---------------- fp32 -> bf16 conversion (4 elems/thread) ----------------
__global__ __launch_bounds__(256) void cvt_f32_bf16_k(const float* __restrict__ in,
                                                      bf16* __restrict__ out, int n4) {
    int i = blockIdx.x * blockDim.x + threadIdx.x;
    if (i >= n4) return;
    float4 v = *(const float4*)(in + (size_t)i * 4);
    union { bf16 h[4]; uint2 u; } o;
    o.h[0] = (bf16)v.x; o.h[1] = (bf16)v.y; o.h[2] = (bf16)v.z; o.h[3] = (bf16)v.w;
    *(uint2*)(out + (size_t)i * 4) = o.u;
}

// ---------------- embedding gather -> x_bf16 [T*B, E] ----------------
__global__ __launch_bounds__(256) void gather_emb_k(const int* __restrict__ inputs,
                                                    const float* __restrict__ emb,
                                                    bf16* __restrict__ xb) {
    int idx = blockIdx.x * blockDim.x + threadIdx.x;  // T*B*E/4 threads
    int m  = idx >> 7;            // row (t*B+b), E/4 = 128 quads per row
    int k  = (idx & 127) << 2;
    int t  = m >> 5;              // B = 32
    int b  = m & 31;
    int tok = inputs[b * kT + t];
    float4 v = *(const float4*)(emb + (size_t)tok * kE + k);
    union { bf16 h[4]; uint2 u; } o;
    o.h[0] = (bf16)v.x; o.h[1] = (bf16)v.y; o.h[2] = (bf16)v.z; o.h[3] = (bf16)v.w;
    *(uint2*)(xb + (size_t)m * kE + k) = o.u;
}

// ---------------- init h0 (bf16) and c (fp32) ----------------
__global__ __launch_bounds__(256) void prep_state_k(const float* __restrict__ hidden,
                                                    const float* __restrict__ cell,
                                                    bf16* __restrict__ h0b,
                                                    float* __restrict__ cbuf) {
    int i = blockIdx.x * blockDim.x + threadIdx.x;   // B*H threads
    h0b[i] = (bf16)hidden[i];
    cbuf[i] = cell[i];
}

// ---------------- input projection GEMM (direct-load, small) ----------------
// C[M,N] = A[M,K] x W[N,K]^T + bias0[n] + bias1[n], row-major C.
// block: 256 thr = 8 waves as 2(M) x 4(N); block tile 64x128; wave tile 32x32.
__global__ __launch_bounds__(256) void gemm_xgates_k(const bf16* __restrict__ A,
                                                     const bf16* __restrict__ W,
                                                     const float* __restrict__ bias0,
                                                     const float* __restrict__ bias1,
                                                     float* __restrict__ C,
                                                     int N, int K) {
    const int tid  = threadIdx.x;
    const int wave = tid >> 5;
    const int lane = tid & 31;
    const int lh   = lane & 15;
    const int kh   = (lane >> 4) << 3;
    const int mblk = blockIdx.y * 64  + (wave >> 2) * 32;
    const int nblk = blockIdx.x * 128 + (wave & 3) * 32;

    const bf16* a0p = A + (size_t)(mblk + lh) * K + kh;
    const bf16* a1p = a0p + (size_t)16 * K;
    const bf16* b0p = W + (size_t)(nblk + lh) * K + kh;
    const bf16* b1p = b0p + (size_t)16 * K;

    v8f acc[2][2];
    acc[0][0] = vzero8(); acc[0][1] = vzero8();
    acc[1][0] = vzero8(); acc[1][1] = vzero8();

    for (int k = 0; k < K; k += 32) {
        FragBF a0, a1, b0, b1;
        a0.q[0] = *(const uint4*)(a0p + k);      a0.q[1] = *(const uint4*)(a0p + k + 16);
        a1.q[0] = *(const uint4*)(a1p + k);      a1.q[1] = *(const uint4*)(a1p + k + 16);
        b0.q[0] = *(const uint4*)(b0p + k);      b0.q[1] = *(const uint4*)(b0p + k + 16);
        b1.q[0] = *(const uint4*)(b1p + k);      b1.q[1] = *(const uint4*)(b1p + k + 16);
        acc[0][0] = wmma_bf16(a0, b0, acc[0][0]);
        acc[0][1] = wmma_bf16(a0, b1, acc[0][1]);
        acc[1][0] = wmma_bf16(a1, b0, acc[1][0]);
        acc[1][1] = wmma_bf16(a1, b1, acc[1][1]);
    }

    const int mlo = (lane >> 4) << 3;
    #pragma unroll
    for (int mi = 0; mi < 2; ++mi) {
        #pragma unroll
        for (int ni = 0; ni < 2; ++ni) {
            const int n = nblk + ni * 16 + lh;
            const float bias = bias0[n] + bias1[n];
            #pragma unroll
            for (int r = 0; r < 8; ++r) {
                const int m = mblk + mi * 16 + mlo + r;
                C[(size_t)m * N + n] = acc[mi][ni][r] + bias;
            }
        }
    }
}

// ---------------- logits GEMM: LDS-staged, async double-buffered ----------------
// out[(b*T+t)*V + n] = sum_k hs[(t*B+b)*H + k] * fcW[n*H + k] + fc_b[n]
// block tile 64(M) x 256(N), 8 waves as 2(M) x 4(N), wave tile 32 x 64.
// Tiles staged to LDS with GLOBAL_LOAD_ASYNC_TO_LDS_B128 (ASYNCcnt), double-buffered.
constexpr int LPAD = 40;    // padded LDS row (bf16 elems): 80B -> bank-conflict-free
__global__ __launch_bounds__(256) void gemm_logits_k(const bf16* __restrict__ A,
                                                     const bf16* __restrict__ W,
                                                     const float* __restrict__ bias,
                                                     float* __restrict__ C) {
    constexpr int K  = kH;       // 1024
    constexpr int NK = K / 32;   // 32 k-steps
    __shared__ bf16 shA[2][64  * LPAD];   // 10.0 KB
    __shared__ bf16 shB[2][256 * LPAD];   // 40.0 KB

    const int tid  = threadIdx.x;
    const int wave = tid >> 5;
    const int lane = tid & 31;
    const int lh   = lane & 15;
    const int kh   = (lane >> 4) << 3;
    const int mblk = blockIdx.y * 64;
    const int nblk = blockIdx.x * 256;
    const int wm   = (wave >> 2) * 32;    // wave M offset within block tile
    const int wn   = (wave & 3)  * 64;    // wave N offset within block tile

    // cooperative copy assignment: 16B chunk id -> (row, 16B col)
    const int crow = tid >> 2;            // 0..63
    const int ckc  = tid & 3;             // 0..3 (4 x 16B = 64B = 32 bf16 per row)

    v8f acc[2][4];
    #pragma unroll
    for (int mi = 0; mi < 2; ++mi)
        #pragma unroll
        for (int ni = 0; ni < 4; ++ni) acc[mi][ni] = vzero8();

    auto issue_tile = [&](int buf, int k0) {
        // A tile: 64 rows x 64B  -> 1 async op per thread
        async_copy_b128(lds_addr_of(&shA[buf][crow * LPAD]) + ckc * 16,
                        A + (size_t)(mblk + crow) * K + k0 + ckc * 8);
        // B tile: 256 rows x 64B -> 4 async ops per thread
        #pragma unroll
        for (int j = 0; j < 4; ++j) {
            const int row = crow + 64 * j;
            async_copy_b128(lds_addr_of(&shB[buf][row * LPAD]) + ckc * 16,
                            W + (size_t)(nblk + row) * K + k0 + ckc * 8);
        }
    };

    issue_tile(0, 0);
    wait_asynccnt0();
    __syncthreads();

    for (int ks = 0; ks < NK; ++ks) {
        const int cur = ks & 1;
        if (ks + 1 < NK) issue_tile(cur ^ 1, (ks + 1) * 32);   // overlap with compute

        FragBF a[2], b[4];
        #pragma unroll
        for (int mi = 0; mi < 2; ++mi) {
            const bf16* p = &shA[cur][(wm + mi * 16 + lh) * LPAD + kh];
            a[mi].q[0] = *(const uint4*)p;
            a[mi].q[1] = *(const uint4*)(p + 16);
        }
        #pragma unroll
        for (int ni = 0; ni < 4; ++ni) {
            const bf16* p = &shB[cur][(wn + ni * 16 + lh) * LPAD + kh];
            b[ni].q[0] = *(const uint4*)p;
            b[ni].q[1] = *(const uint4*)(p + 16);
        }
        #pragma unroll
        for (int mi = 0; mi < 2; ++mi)
            #pragma unroll
            for (int ni = 0; ni < 4; ++ni)
                acc[mi][ni] = wmma_bf16(a[mi], b[ni], acc[mi][ni]);

        wait_asynccnt0();     // next buffer landed (this wave's copies)
        __syncthreads();      // ...and everyone else's; also: all done reading cur
    }

    // epilogue: bias + (t*B+b) -> (b*T+t) remap
    const int mlo = (lane >> 4) << 3;
    #pragma unroll
    for (int mi = 0; mi < 2; ++mi) {
        #pragma unroll
        for (int ni = 0; ni < 4; ++ni) {
            const int n = nblk + wn + ni * 16 + lh;
            const float bv = bias[n];
            #pragma unroll
            for (int r = 0; r < 8; ++r) {
                const int m = mblk + wm + mi * 16 + mlo + r;   // m = t*B + b
                const int t = m >> 5;
                const int b2 = m & 31;
                C[((size_t)b2 * kT + t) * kV + n] = acc[mi][ni][r] + bv;
            }
        }
    }
}

// ---------------- one LSTM time step ----------------
// each wave owns 16 h-columns x all 32 batch rows x all 4 gates ->
// i/f/g/o for the same (b,n) land in the same lane & accumulator slot.
__global__ __launch_bounds__(256) void lstm_step_k(const bf16* __restrict__ hprev,  // [B,H] bf16
                                                   const float* __restrict__ xg,    // [B,4H] (this t)
                                                   const bf16* __restrict__ Whh,    // [4H,H] bf16
                                                   float* __restrict__ cbuf,        // [B,H]
                                                   bf16* __restrict__ hout) {       // [B,H] bf16 (= hs[t])
    const int tid   = threadIdx.x;
    const int wave  = tid >> 5;
    const int lane  = tid & 31;
    const int lh    = lane & 15;
    const int kh    = (lane >> 4) << 3;
    const int nbase = (blockIdx.x * 8 + wave) * 16;   // 64 waves x 16 = H

    const bf16* a0p = hprev + (size_t)lh * kH + kh;
    const bf16* a1p = a0p + (size_t)16 * kH;
    const bf16* bp[4];
    #pragma unroll
    for (int g = 0; g < 4; ++g)
        bp[g] = Whh + (size_t)(g * kH + nbase + lh) * kH + kh;

    v8f acc[4][2];
    #pragma unroll
    for (int g = 0; g < 4; ++g) { acc[g][0] = vzero8(); acc[g][1] = vzero8(); }

    for (int k = 0; k < kH; k += 32) {
        FragBF a0, a1;
        a0.q[0] = *(const uint4*)(a0p + k);  a0.q[1] = *(const uint4*)(a0p + k + 16);
        a1.q[0] = *(const uint4*)(a1p + k);  a1.q[1] = *(const uint4*)(a1p + k + 16);
        #pragma unroll
        for (int g = 0; g < 4; ++g) {
            FragBF b;
            b.q[0] = *(const uint4*)(bp[g] + k);  b.q[1] = *(const uint4*)(bp[g] + k + 16);
            acc[g][0] = wmma_bf16(a0, b, acc[g][0]);
            acc[g][1] = wmma_bf16(a1, b, acc[g][1]);
        }
    }

    const int n   = nbase + lh;
    const int mlo = (lane >> 4) << 3;
    #pragma unroll
    for (int mi = 0; mi < 2; ++mi) {
        #pragma unroll
        for (int r = 0; r < 8; ++r) {
            const int m = mi * 16 + mlo + r;            // batch row
            const float* xgp = xg + (size_t)m * k4H + n;
            const float gi = sigmoidf_(acc[0][mi][r] + xgp[0 * kH]);
            const float gf = sigmoidf_(acc[1][mi][r] + xgp[1 * kH]);
            const float gg = tanhf    (acc[2][mi][r] + xgp[2 * kH]);
            const float go = sigmoidf_(acc[3][mi][r] + xgp[3 * kH]);
            const size_t o = (size_t)m * kH + n;
            const float cn = gf * cbuf[o] + gi * gg;
            cbuf[o] = cn;
            hout[o] = (bf16)(go * tanhf(cn));
        }
    }
}

// ---------------- host-side orchestration ----------------
extern "C" void kernel_launch(void* const* d_in, const int* in_sizes, int n_in,
                              void* d_out, int out_size, void* d_ws, size_t ws_size,
                              hipStream_t stream) {
    const int*   inputs = (const int*)  d_in[0];
    const float* hidden = (const float*)d_in[1];
    const float* cell   = (const float*)d_in[2];
    //            d_in[3] = generate_len (unused; T is static)
    const float* emb    = (const float*)d_in[4];
    const float* Wih    = (const float*)d_in[5];
    const float* Whh    = (const float*)d_in[6];
    const float* bih    = (const float*)d_in[7];
    const float* bhh    = (const float*)d_in[8];
    const float* fcW    = (const float*)d_in[9];
    const float* fcb    = (const float*)d_in[10];
    float*       out    = (float*)d_out;

    // workspace carve-up (all 256B aligned)
    char* ws = (char*)d_ws;
    size_t off = 0;
    float* xg      = (float*)(ws + off); off += (size_t)kT * kB * k4H * 4;  // 33.5 MB
    bf16*  hs      = (bf16*) (ws + off); off += (size_t)kT * kB * kH  * 2;  //  4.2 MB
    bf16*  h0b     = (bf16*) (ws + off); off += (size_t)kB * kH * 2;        //  64 KB
    float* cbuf    = (float*)(ws + off); off += (size_t)kB * kH * 4;        // 128 KB
    bf16*  wih_b   = (bf16*) (ws + off); off += (size_t)k4H * kE * 2;       //  4.2 MB
    bf16*  whh_b   = (bf16*) (ws + off); off += (size_t)k4H * kH * 2;       //  8.4 MB
    bf16*  fcw_b   = (bf16*) (ws + off); off += (size_t)kV  * kH * 2;       // 65.5 MB
    bf16*  xb      = (bf16*) (ws + off); off += (size_t)kT * kB * kE * 2;   //  2.1 MB

    // 1) weight conversions to bf16
    {
        int n4 = (k4H * kE) / 4;
        cvt_f32_bf16_k<<<(n4 + 255) / 256, 256, 0, stream>>>(Wih, wih_b, n4);
    }
    {
        int n4 = (k4H * kH) / 4;
        cvt_f32_bf16_k<<<(n4 + 255) / 256, 256, 0, stream>>>(Whh, whh_b, n4);
    }
    {
        int n4 = (kV * kH) / 4;
        cvt_f32_bf16_k<<<(n4 + 255) / 256, 256, 0, stream>>>(fcW, fcw_b, n4);
    }

    // 2) embedding gather (time-major rows m = t*B+b)
    {
        int nthr = kT * kB * kE / 4;
        gather_emb_k<<<nthr / 256, 256, 0, stream>>>(inputs, emb, xb);
    }

    // 3) state init
    prep_state_k<<<(kB * kH) / 256, 256, 0, stream>>>(hidden, cell, h0b, cbuf);

    // 4) input projection: x_gates = xb @ W_ih^T + b_ih + b_hh   [2048 x 4096]
    {
        dim3 grid(k4H / 128, (kT * kB) / 64);
        gemm_xgates_k<<<grid, 256, 0, stream>>>(xb, wih_b, bih, bhh, xg, k4H, kE);
    }

    // 5) recurrent scan: 64 dependent steps (stream order = dependency order)
    for (int t = 0; t < kT; ++t) {
        const bf16* hp = (t == 0) ? h0b : (hs + (size_t)(t - 1) * kB * kH);
        lstm_step_k<<<8, 256, 0, stream>>>(hp,
                                           xg + (size_t)t * kB * k4H,
                                           whh_b, cbuf,
                                           hs + (size_t)t * kB * kH);
    }

    // 6) logits: out[b,t,:] = hs[t,b,:] @ fc_W^T + fc_b   [2048 x 32000]
    {
        dim3 grid(kV / 256, (kT * kB) / 64);
        gemm_logits_k<<<grid, 256, 0, stream>>>(hs, fcw_b, fcb, out);
    }
}
